// NTXentLoss_43989055045816
// MI455X (gfx1250) — matrix operand order repaired and checked by reading, not compiled, not run on hardware
//
#include <hip/hip_runtime.h>
#include <hip/hip_bf16.h>

// ---------------------------------------------------------------------------
// NT-Xent loss, fused for MI455X (gfx1250, wave32, WMMA 16x16x32 bf16).
// sim matrix (8192x8192) is never materialized: row-sums of exp(sim/tau)
// are accumulated tile-by-tile in registers out of v_wmma_f32_16x16x32_bf16.
// fp32 accuracy recovered via bf16 hi/lo split (3 WMMAs per K-chunk).
// A tile staged in LDS in fragment order (shared by all 8 waves) to keep
// VGPR pressure low -> no scratch spills in the hot loop.
// ---------------------------------------------------------------------------

#define BSZ   4096
#define DIM   256
#define NROWS 8192            // 2*BSZ
#define INV_TAU 2.0f          // 1/0.5
#define EPSV 1e-8f

typedef __bf16 v16bf __attribute__((ext_vector_type(16)));
typedef float  v8f   __attribute__((ext_vector_type(8)));
typedef unsigned short su8 __attribute__((ext_vector_type(8)));
typedef unsigned short u16;

union Frag {                  // one WMMA 16-bit operand (8 VGPRs / lane)
    v16bf bf;
    su8   h[2];
};

__device__ __forceinline__ u16 f32_to_bf16_rne(float f) {
    unsigned int u = __float_as_uint(f);
    return (u16)((u + 0x7FFFu + ((u >> 16) & 1u)) >> 16);
}
__device__ __forceinline__ float bf16_bits_to_f32(u16 h) {
    return __uint_as_float(((unsigned int)h) << 16);
}

// -------- kernel 1: normalize rows, split into bf16 hi/lo, also transpose ---
__global__ __launch_bounds__(DIM) void ntx_normalize(
                              const float* __restrict__ zi,
                              const float* __restrict__ zj,
                              u16* __restrict__ zhi, u16* __restrict__ zlo,
                              u16* __restrict__ zthi, u16* __restrict__ ztlo,
                              float* __restrict__ scl) {
    const int r = blockIdx.x;           // 0..8191
    const int k = threadIdx.x;          // 0..255
    const float* src = (r < BSZ) ? (zi + (size_t)r * DIM)
                                 : (zj + (size_t)(r - BSZ) * DIM);
    float v = src[k];

    __shared__ float red[DIM];
    red[k] = v * v;
    __syncthreads();
    for (int s = DIM / 2; s > 0; s >>= 1) {
        if (k < s) red[k] += red[k + s];
        __syncthreads();
    }
    const float sc = 1.0f / fmaxf(sqrtf(red[0]), 1e-8f);   // torch norm clamp
    if (k == 0) scl[r] = sc;

    const float zv = v * sc;
    const u16 hb = f32_to_bf16_rne(zv);
    const u16 lb = f32_to_bf16_rne(zv - bf16_bits_to_f32(hb));

    zhi[(size_t)r * DIM + k] = hb;           // row-major (A operand)
    zlo[(size_t)r * DIM + k] = lb;
    zthi[(size_t)k * NROWS + r] = hb;        // transposed (not read by ntx_gemm_rowsum)
    ztlo[(size_t)k * NROWS + r] = lb;
}

// -------- kernel 2: fused zn*zn^T -> exp -> row sums (skip diagonal) --------
// grid: 512 blocks (16 rows each), 256 threads = 8 waves striping 512 col blks
__global__ __launch_bounds__(256, 1) void ntx_gemm_rowsum(
                                const u16* __restrict__ zhi,
                                const u16* __restrict__ zlo,
                                const u16* __restrict__ zthi,
                                const u16* __restrict__ ztlo,
                                float* __restrict__ rowsum) {
    const int tid  = threadIdx.x;
    const int lane = tid & 31;
    const int wid  = tid >> 5;        // 0..7
    const int hh   = lane >> 4;       // lane half: 0/1
    const int m    = lane & 15;
    const int rowBase = blockIdx.x * 16;

    // A tile (16 rows x 256 K, hi+lo) staged in LDS in WMMA-fragment order:
    // [chunk][lane][16 halves]; lane m holds K{0..7,16..23}+32c,
    // lane m+16 holds K{8..15,24..31}+32c.  Shared by all 8 waves.
    __shared__ __align__(16) u16 AhiL[8][32][16];
    __shared__ __align__(16) u16 AloL[8][32][16];
    {
        const int c0 = tid >> 5;          // chunk this thread fills
        const int l0 = tid & 31;          // fragment lane it fills
        const int h0 = l0 >> 4;
        const int m0 = l0 & 15;
        const u16* arH = zhi + (size_t)(rowBase + m0) * DIM;
        const u16* arL = zlo + (size_t)(rowBase + m0) * DIM;
        const int k0 = 32 * c0 + 8 * h0;
        *(su8*)&AhiL[c0][l0][0] = *(const su8*)(arH + k0);
        *(su8*)&AhiL[c0][l0][8] = *(const su8*)(arH + k0 + 16);
        *(su8*)&AloL[c0][l0][0] = *(const su8*)(arL + k0);
        *(su8*)&AloL[c0][l0][8] = *(const su8*)(arL + k0 + 16);
    }
    __syncthreads();

    float rs[8];
#pragma unroll
    for (int v = 0; v < 8; ++v) rs[v] = 0.0f;

    for (int cb = wid; cb < NROWS / 16; cb += 8) {
        v8f acc = {};
#pragma unroll
        for (int c = 0; c < 8; ++c) {
            Frag Ahi, Alo, Bhi, Blo;
            // A fragments from LDS (ds_load_b128, conflict-free)
            Ahi.h[0] = *(const su8*)&AhiL[c][lane][0];
            Ahi.h[1] = *(const su8*)&AhiL[c][lane][8];
            Alo.h[0] = *(const su8*)&AloL[c][lane][0];
            Alo.h[1] = *(const su8*)&AloL[c][lane][8];
            // B operand from L2: lane m = N col (row cb*16+m of zn), halves = K{0..7,16..23}+8hh+32c.
            const size_t boff = (size_t)(cb * 16 + m) * DIM + 32 * c + 8 * hh;
            Bhi.h[0] = *(const su8*)(zhi + boff);
            Bhi.h[1] = *(const su8*)(zhi + boff + 16);
            Blo.h[0] = *(const su8*)(zlo + boff);
            Blo.h[1] = *(const su8*)(zlo + boff + 16);
            // (Ahi+Alo)(Bhi+Blo) ~= AhiBhi + AhiBlo + AloBhi  (fp32 accum)
            acc = __builtin_amdgcn_wmma_f32_16x16x32_bf16(false, Ahi.bf,
                      false, Bhi.bf, (short)0, acc, false, false);
            acc = __builtin_amdgcn_wmma_f32_16x16x32_bf16(false, Ahi.bf,
                      false, Blo.bf, (short)0, acc, false, false);
            acc = __builtin_amdgcn_wmma_f32_16x16x32_bf16(false, Alo.bf,
                      false, Bhi.bf, (short)0, acc, false, false);
        }
        // C layout: lane -> N = m, VGPR v -> M = v + 8*hh
        const int gnBase = cb * 16;
#pragma unroll
        for (int v = 0; v < 8; ++v) {
            const int gm = rowBase + v + 8 * hh;
            const int gn = gnBase + m;
            const float e = __expf(acc[v] * INV_TAU);
            rs[v] += (gm == gn) ? 0.0f : e;   // negatives exclude diagonal
        }
    }

    // reduce over the 16 lanes of each half (sum over N within this wave)
#pragma unroll
    for (int v = 0; v < 8; ++v) {
        float x = rs[v];
        x += __shfl_xor(x, 1, 16);
        x += __shfl_xor(x, 2, 16);
        x += __shfl_xor(x, 4, 16);
        x += __shfl_xor(x, 8, 16);
        rs[v] = x;
    }

    __shared__ float part[8][16];
    if (m == 0) {
#pragma unroll
        for (int v = 0; v < 8; ++v) part[wid][8 * hh + v] = rs[v];
    }
    __syncthreads();
    if (tid < 16) {                       // deterministic fixed-order combine
        float s = 0.0f;
#pragma unroll
        for (int w = 0; w < 8; ++w) s += part[w][tid];
        rowsum[rowBase + tid] = s;        // '=' write: no init required
    }
}

// -------- kernel 3: exact fp32 positive-pair sims + per-row loss term ------
// grid: 1024 blocks x 256 threads; one wave per row.
__global__ __launch_bounds__(256) void ntx_row_terms(
                              const float* __restrict__ zi,
                              const float* __restrict__ zj,
                              const float* __restrict__ scl,
                              const float* __restrict__ rowsum,
                              float* __restrict__ terms) {
    const int wid  = threadIdx.x >> 5;
    const int lane = threadIdx.x & 31;
    const int i = blockIdx.x * 8 + wid;
    const int p = (i < BSZ) ? i + BSZ : i - BSZ;
    const float* a = (i < BSZ) ? zi + (size_t)i * DIM : zj + (size_t)(i - BSZ) * DIM;
    const float* b = (p < BSZ) ? zi + (size_t)p * DIM : zj + (size_t)(p - BSZ) * DIM;

    float dot = 0.0f;
#pragma unroll
    for (int j = 0; j < DIM; j += 32) dot = fmaf(a[lane + j], b[lane + j], dot);
    dot += __shfl_xor(dot, 1, 32);
    dot += __shfl_xor(dot, 2, 32);
    dot += __shfl_xor(dot, 4, 32);
    dot += __shfl_xor(dot, 8, 32);
    dot += __shfl_xor(dot, 16, 32);

    if (lane == 0) {
        const float sim_pos = dot * scl[i] * scl[p] * INV_TAU;
        // -log(exp(sim_pos)/(negsum+eps)) == log(negsum+eps) - sim_pos
        terms[i] = logf(rowsum[i] + EPSV) - sim_pos;
    }
}

// -------- kernel 4: final mean over 8192 terms (single block, fixed order) --
__global__ __launch_bounds__(256) void ntx_reduce(
                           const float* __restrict__ terms,
                           float* __restrict__ out) {
    __shared__ float red[256];
    const int t = threadIdx.x;
    float s = 0.0f;
    for (int j = t; j < NROWS; j += 256) s += terms[j];
    red[t] = s;
    __syncthreads();
    for (int st = 128; st > 0; st >>= 1) {
        if (t < st) red[t] += red[t + st];
        __syncthreads();
    }
    if (t == 0) out[0] = red[0] / (float)NROWS;
}

extern "C" void kernel_launch(void* const* d_in, const int* in_sizes, int n_in,
                              void* d_out, int out_size, void* d_ws, size_t ws_size,
                              hipStream_t stream) {
    const float* zi = (const float*)d_in[0];
    const float* zj = (const float*)d_in[1];

    char* ws = (char*)d_ws;
    const size_t ZSZ = (size_t)NROWS * DIM * sizeof(u16);     // 4 MB each
    u16*   zhi    = (u16*)(ws + 0 * ZSZ);
    u16*   zlo    = (u16*)(ws + 1 * ZSZ);
    u16*   zthi   = (u16*)(ws + 2 * ZSZ);
    u16*   ztlo   = (u16*)(ws + 3 * ZSZ);
    float* rowsum = (float*)(ws + 4 * ZSZ);
    float* terms  = (float*)(ws + 4 * ZSZ + (size_t)NROWS * 4);
    float* scl    = (float*)(ws + 4 * ZSZ + (size_t)NROWS * 8);

    ntx_normalize<<<NROWS, DIM, 0, stream>>>(zi, zj, zhi, zlo, zthi, ztlo, scl);
    ntx_gemm_rowsum<<<NROWS / 16, 256, 0, stream>>>(zhi, zlo, zthi, ztlo, rowsum);
    ntx_row_terms<<<NROWS / 8, 256, 0, stream>>>(zi, zj, scl, rowsum, terms);
    ntx_reduce<<<1, 256, 0, stream>>>(terms, (float*)d_out);
}
